// RingDilatedAttentionHybrid_23029614641318
// MI455X (gfx1250) — matrix operand order repaired and checked by reading, not compiled, not run on hardware
//
#include <hip/hip_runtime.h>

typedef __attribute__((ext_vector_type(16))) __bf16 v16bf;
typedef __attribute__((ext_vector_type(8)))  float  v8f;

#define H_TOT   12
#define D_HEAD  64
#define M_DIL   2048
#define LDSTR   36   // dwords per 64-halfword row: keeps 16B alignment for b128 LDS
                     // loads AND stride 36 mod 64 banks is conflict-free over 16 lanes

// Native bf16 conversion (v_cvt_pk_bf16_f32 on gfx1250)
__device__ __forceinline__ unsigned pack_bf16(float lo, float hi) {
  union { __bf16 h[2]; unsigned u; } r;
  r.h[0] = (__bf16)lo;
  r.h[1] = (__bf16)hi;
  return r.u;
}

// ---- 16-lane xor butterfly step, VALU-only via v_permlane16_b32 ----
constexpr unsigned mksel(int base, int m) {
  unsigned s = 0;
  for (int j = 0; j < 8; ++j) s |= (unsigned)(((base + j) ^ m) & 0xF) << (4 * j);
  return s;
}
template <int M>
__device__ __forceinline__ float xor16(float x) {
#if __has_builtin(__builtin_amdgcn_permlane16)
  constexpr unsigned s0 = mksel(0, M);
  constexpr unsigned s1 = mksel(8, M);
  int r = __builtin_amdgcn_permlane16(__float_as_int(x), __float_as_int(x),
                                      (int)s0, (int)s1, false, false);
  return __int_as_float(r);
#else
  int r = __builtin_amdgcn_ds_swizzle(__float_as_int(x), (M << 10) | 0x1F);
  return __int_as_float(r);
#endif
}
__device__ __forceinline__ float rowmax16(float x) {
  x = fmaxf(x, xor16<1>(x));
  x = fmaxf(x, xor16<2>(x));
  x = fmaxf(x, xor16<4>(x));
  x = fmaxf(x, xor16<8>(x));
  return x;
}
__device__ __forceinline__ float rowsum16(float x) {
  x += xor16<1>(x);
  x += xor16<2>(x);
  x += xor16<4>(x);
  x += xor16<8>(x);
  return x;
}

// A-operand (16x32 bf16, MxK): lane = M (lane%16); dword v holds K pair:
// K = koff + (v<4?0:16) + 8*half + 2*(v&3)  ->  dwords = base, base+1..3 and base+8..11
__device__ __forceinline__ v16bf ld_opA(const unsigned* base_dw, int prim, int half, int koff) {
  union { v16bf v; uint4 q[2]; } r;
  const unsigned* p = base_dw + prim * LDSTR + (koff >> 1) + half * 4;
  r.q[0] = *(const uint4*)(p);       // vv = 0..3
  r.q[1] = *(const uint4*)(p + 8);   // vv = 4..7
  return r.v;
}
// B-operand (32x16 bf16, KxN): lane = N (lane%16); dword v holds K pair:
// K = koff + (v<4?0:16) + 8*half + 2*(v&3)  ->  dwords base..base+3 and base+8..11
__device__ __forceinline__ v16bf ld_opB(const unsigned* base_dw, int prim, int half, int koff) {
  union { v16bf v; uint4 q[2]; } r;
  const unsigned* p = base_dw + prim * LDSTR + (koff >> 1) + half * 4;
  r.q[0] = *(const uint4*)(p);
  r.q[1] = *(const uint4*)(p + 8);
  return r.v;
}

__global__ void zero_out_kernel(float* __restrict__ out, int n) {
  int i = blockIdx.x * blockDim.x + threadIdx.x;
  int stride = gridDim.x * blockDim.x;
  for (; i < n; i += stride) out[i] = 0.0f;
}

__global__ __launch_bounds__(256) void dilated_attn_kernel(
    const float* __restrict__ q, const float* __restrict__ k,
    const float* __restrict__ v, const int* __restrict__ pcausal,
    float* __restrict__ out) {
  // LDS: K chunk [64][64]bf16, V^T chunk [64][64]bf16, per-wave P scratch [16][64]bf16
  __shared__ __align__(16) unsigned smem[2 * 64 * LDSTR + 8 * 16 * LDSTR];
  unsigned* Klds  = smem;
  unsigned* VTlds = smem + 64 * LDSTR;

  const int lane = threadIdx.x & 31;
  const int wave = threadIdx.x >> 5;
  const int half = lane >> 4;
  const int ln   = lane & 15;

  // Block decode: 448 blocks = 28 units x 16 q-blocks
  int bid = blockIdx.x;
  int qb  = bid & 15;
  int u   = bid >> 4;
  int g, ul;
  if (u < 16)      { g = 0; ul = u; }
  else if (u < 24) { g = 1; ul = u - 16; }
  else             { g = 2; ul = u - 24; }
  const int nseg = 4 >> g;                  // 4,2,1 segments
  const int head = g * 4 + (ul / nseg);
  const int seg  = ul % nseg;
  const int slen = 2048 << g;
  const int rdil = 1 << g;
  const int offs = g & (rdil - 1);          // i % r -> 0,1,2
  const int segtok = seg * slen + offs;     // dilated j -> global token segtok + j*rdil

  const int causal = *pcausal;
  const int qbase  = qb * 128;

  // 1/sqrt(64) * log2(e): folded into Q so WMMA emits log2-domain scores directly
  const float kscale = 0.125f * 1.44269504088896340736f;

  // ---- Load Q tile (16 rows x d=64) straight into two A-operands (bf16, pre-scaled) ----
  const int qj_lane = qbase + wave * 16 + ln;
  const float* qrow = q + ((size_t)(segtok + qj_lane * rdil) * H_TOT + head) * D_HEAD;
  v16bf qa0, qa1;
  {
    union { v16bf v; unsigned u8[8]; } t0, t1;
#pragma unroll
    for (int vv = 0; vv < 8; ++vv) {
      int kb = ((vv & 4) ? 16 : 0) + half * 8 + (vv & 3) * 2;
      t0.u8[vv] = pack_bf16(qrow[kb] * kscale,      qrow[kb + 1] * kscale);
      t1.u8[vv] = pack_bf16(qrow[kb + 32] * kscale, qrow[kb + 33] * kscale);
    }
    qa0 = t0.v; qa1 = t1.v;
  }

  v8f Oacc[4];
#pragma unroll
  for (int t = 0; t < 4; ++t) Oacc[t] = v8f{0, 0, 0, 0, 0, 0, 0, 0};
  float mrun[8], lrun[8];
#pragma unroll
  for (int rr = 0; rr < 8; ++rr) { mrun[rr] = -__builtin_inff(); lrun[rr] = 0.0f; }

  const int nch = causal ? (qb * 2 + 2) : (M_DIL / 64);

  unsigned* Pl = smem + 2 * 64 * LDSTR + wave * 16 * LDSTR;
  __bf16*   Pb = (__bf16*)Pl;

  for (int ch = 0; ch < nch; ++ch) {
    const int kvbase = ch * 64;
    __syncthreads();
    // ---- cooperative load of K (row-major) and V^T into LDS as bf16 ----
#pragma unroll
    for (int it = 0; it < 4; ++it) {
      int t   = threadIdx.x + it * 256;
      int tok = t >> 4;                      // 0..63
      int f4  = t & 15;                      // float4 column
      const size_t gtok = (size_t)(segtok + (kvbase + tok) * rdil);
      float4 k4 = *(const float4*)(k + (gtok * H_TOT + head) * D_HEAD + f4 * 4);
      float4 v4 = *(const float4*)(v + (gtok * H_TOT + head) * D_HEAD + f4 * 4);
      uint2 kk;
      kk.x = pack_bf16(k4.x, k4.y);
      kk.y = pack_bf16(k4.z, k4.w);
      *(uint2*)(Klds + tok * LDSTR + f4 * 2) = kk;     // ds_store_b64
      __bf16* vt = (__bf16*)VTlds;
      vt[(f4 * 4 + 0) * (2 * LDSTR) + tok] = (__bf16)v4.x;
      vt[(f4 * 4 + 1) * (2 * LDSTR) + tok] = (__bf16)v4.y;
      vt[(f4 * 4 + 2) * (2 * LDSTR) + tok] = (__bf16)v4.z;
      vt[(f4 * 4 + 3) * (2 * LDSTR) + tok] = (__bf16)v4.w;
      if (ch + 1 < nch) {  // prefetch next chunk into cache (global_prefetch_b8)
        const float* kn = k + ((size_t)(segtok + (kvbase + 64 + tok) * rdil) * H_TOT + head) * D_HEAD + f4 * 4;
        __builtin_prefetch(kn, 0, 0);
      }
    }
    __syncthreads();

    // ---- S = Q K^T (4 column tiles x 2 K-steps of 32); already in log2 units ----
    v8f S[4];
#pragma unroll
    for (int t = 0; t < 4; ++t) {
      v8f c = v8f{0, 0, 0, 0, 0, 0, 0, 0};
      v16bf b0 = ld_opB(Klds, t * 16 + ln, half, 0);
      c = __builtin_amdgcn_wmma_f32_16x16x32_bf16(false, qa0, false, b0, (short)0, c, false, false);
      v16bf b1 = ld_opB(Klds, t * 16 + ln, half, 32);
      c = __builtin_amdgcn_wmma_f32_16x16x32_bf16(false, qa1, false, b1, (short)0, c, false, false);
      S[t] = c;
    }

    // ---- causal mask: only the diagonal chunks need it (uniform per-wave branch) ----
    const bool need_mask = causal && (kvbase + 63 > qbase + wave * 16);
    if (need_mask) {
#pragma unroll
      for (int t = 0; t < 4; ++t) {
        int kj = kvbase + t * 16 + ln;
#pragma unroll
        for (int rr = 0; rr < 8; ++rr) {
          int qj = qbase + wave * 16 + rr + 8 * half;
          if (kj > qj) S[t][rr] = -__builtin_inff();
        }
      }
    }

    // ---- row max over the 16-lane column stripe (VALU permlane butterflies) ----
    float rmax[8];
#pragma unroll
    for (int rr = 0; rr < 8; ++rr) {
      float x = fmaxf(fmaxf(S[0][rr], S[1][rr]), fmaxf(S[2][rr], S[3][rr]));
      rmax[rr] = rowmax16(x);
    }

    float corr[8], rsum[8];
#pragma unroll
    for (int rr = 0; rr < 8; ++rr) {
      float mnew = fmaxf(mrun[rr], rmax[rr]);
      corr[rr] = exp2f(mrun[rr] - mnew);
      mrun[rr] = mnew;
    }
#pragma unroll
    for (int t = 0; t < 4; ++t)
#pragma unroll
      for (int rr = 0; rr < 8; ++rr)
        S[t][rr] = exp2f(S[t][rr] - mrun[rr]);
#pragma unroll
    for (int rr = 0; rr < 8; ++rr) {
      float x = (S[0][rr] + S[1][rr]) + (S[2][rr] + S[3][rr]);
      rsum[rr] = rowsum16(x);
      lrun[rr] = lrun[rr] * corr[rr] + rsum[rr];
#pragma unroll
      for (int t = 0; t < 4; ++t) Oacc[t][rr] *= corr[rr];
    }

    // ---- P (C layout) -> bf16 per-wave LDS scratch -> A layout ----
#pragma unroll
    for (int t = 0; t < 4; ++t)
#pragma unroll
      for (int rr = 0; rr < 8; ++rr)
        Pb[(rr + 8 * half) * (2 * LDSTR) + t * 16 + ln] = (__bf16)S[t][rr];
    // same-wave LDS ops are in-order; compiler inserts the DScnt waits
    v16bf pa0 = ld_opA(Pl, ln, half, 0);
    v16bf pa1 = ld_opA(Pl, ln, half, 32);

    // ---- O += P V (4 d-tiles x 2 K-steps of 32 over the 64 kv tokens) ----
#pragma unroll
    for (int t = 0; t < 4; ++t) {
      v16bf vb0 = ld_opB(VTlds, t * 16 + ln, half, 0);
      Oacc[t] = __builtin_amdgcn_wmma_f32_16x16x32_bf16(false, pa0, false, vb0, (short)0, Oacc[t], false, false);
      v16bf vb1 = ld_opB(VTlds, t * 16 + ln, half, 32);
      Oacc[t] = __builtin_amdgcn_wmma_f32_16x16x32_bf16(false, pa1, false, vb1, (short)0, Oacc[t], false, false);
    }
  }

  // ---- normalize and write (fp32) to the dilated rows ----
#pragma unroll
  for (int rr = 0; rr < 8; ++rr) {
    float inv = 1.0f / lrun[rr];
    int qj = qbase + wave * 16 + rr + 8 * half;
    float* op = out + ((size_t)(segtok + qj * rdil) * H_TOT + head) * D_HEAD;
#pragma unroll
    for (int t = 0; t < 4; ++t)
      op[t * 16 + ln] = Oacc[t][rr] * inv;
  }
}

extern "C" void kernel_launch(void* const* d_in, const int* in_sizes, int n_in,
                              void* d_out, int out_size, void* d_ws, size_t ws_size,
                              hipStream_t stream) {
  (void)in_sizes; (void)n_in; (void)d_ws; (void)ws_size;
  const float* q  = (const float*)d_in[0];
  const float* k  = (const float*)d_in[1];
  const float* v  = (const float*)d_in[2];
  const int*   pc = (const int*)d_in[3];
  float* out = (float*)d_out;

  // zero-fill (non-dilated rows of groups 1 and 2 must be zero)
  zero_out_kernel<<<2048, 256, 0, stream>>>(out, out_size);
  // 28 units x 16 q-blocks of 128 rows
  dilated_attn_kernel<<<448, 256, 0, stream>>>(q, k, v, pc, out);
}
